// SHSA_14285061226833
// MI455X (gfx1250) — matrix operand order repaired and checked by reading, not compiled, not run on hardware
//
#include <hip/hip_runtime.h>

#define DIMC 384
#define PDIMC 96
#define QKC 16
#define QKVOUT 128
#define NBATCH 64
#define NTOK 1024
#define EPSV 1e-3f

typedef _Float16 half8  __attribute__((ext_vector_type(8)));
typedef _Float16 half16 __attribute__((ext_vector_type(16)));
typedef float    float8 __attribute__((ext_vector_type(8)));

__device__ __forceinline__ void lds_fence() {
    asm volatile("s_wait_dscnt 0" ::: "memory");
}

__device__ __forceinline__ float8 fzero8() {
    float8 z;
#pragma unroll
    for (int j = 0; j < 8; ++j) z[j] = 0.0f;
    return z;
}

// ---------------------------------------------------------------------------
// Prep: fuse BN into qkv weights, fold attn scale (0.25) into q columns,
// and write weights in WMMA B-fragment order: [kc][nt][lane][e] halves.
// ---------------------------------------------------------------------------
__global__ void prep_qkv(const float* __restrict__ w, const float* __restrict__ bias,
                         const float* __restrict__ g, const float* __restrict__ be,
                         const float* __restrict__ mean, const float* __restrict__ var,
                         _Float16* __restrict__ wqB, float* __restrict__ bq) {
    int i = blockIdx.x * 256 + threadIdx.x;
    if (i < QKVOUT) {
        float gp = g[i] * rsqrtf(var[i] + EPSV);
        float bf = (bias[i] - mean[i]) * gp + be[i];
        if (i < QKC) bf *= 0.25f;
        bq[i] = bf;
    }
    if (i >= PDIMC * QKVOUT) return;
    int c = i / QKVOUT, f = i % QKVOUT;
    float gp = g[f] * rsqrtf(var[f] + EPSV);
    float wf = w[i] * gp;
    if (f < QKC) wf *= 0.25f;
    int kc = c >> 5, kk = c & 31, hh = kk >> 4, e = kk & 15;
    int nt = f >> 4, n = f & 15;
    int lane = hh * 16 + n;
    wqB[(size_t)((kc * 8 + nt) * 32 + lane) * 16 + e] = (_Float16)wf;
}

__global__ void prep_proj(const float* __restrict__ w, const float* __restrict__ bias,
                          const float* __restrict__ g, const float* __restrict__ be,
                          const float* __restrict__ mean, const float* __restrict__ var,
                          _Float16* __restrict__ wpB, float* __restrict__ bp) {
    int i = blockIdx.x * 256 + threadIdx.x;
    if (i < DIMC) {
        float gp = g[i] * rsqrtf(var[i] + EPSV);
        bp[i] = (bias[i] - mean[i]) * gp + be[i];
    }
    if (i >= DIMC * DIMC) return;
    int c = i / DIMC, f = i % DIMC;
    float gp = g[f] * rsqrtf(var[f] + EPSV);
    float wf = w[i] * gp;
    int kc = c >> 5, kk = c & 31, hh = kk >> 4, e = kk & 15;
    int nt = f >> 4, n = f & 15;
    int lane = hh * 16 + n;
    wpB[(size_t)((kc * 24 + nt) * 32 + lane) * 16 + e] = (_Float16)wf;
}

// ---------------------------------------------------------------------------
// Kernel 1: LayerNorm(x[:, :96]) + fused QKV GEMM (K=96, N=128) per 16-row
// tile per wave. Writes q, k (f16 [row][16]) and v directly in B-fragment
// layout vB[b][kc32][t6][lane][e].
// ---------------------------------------------------------------------------
__global__ __launch_bounds__(128) void k_qkv(
    const float* __restrict__ x, const float* __restrict__ lng,
    const float* __restrict__ lnb, const _Float16* __restrict__ wqB,
    const float* __restrict__ bq, _Float16* __restrict__ qw,
    _Float16* __restrict__ kw, _Float16* __restrict__ vB) {
    __shared__ __align__(16) _Float16 Abuf[4][16 * 96];
    int wave = threadIdx.x >> 5;
    int lane = threadIdx.x & 31;
    int row0 = (blockIdx.x * 4 + wave) * 16;

    // LayerNorm: two lanes per row, 48 cols each.
    {
        int r = lane >> 1, part = lane & 1;
        const float* xr = x + (size_t)(row0 + r) * DIMC + part * 48;
        float s = 0.f, s2 = 0.f;
        float vals[48];
#pragma unroll
        for (int j = 0; j < 48; ++j) {
            float v = xr[j];
            vals[j] = v; s += v; s2 += v * v;
        }
        s  += __shfl_xor(s, 1, 32);
        s2 += __shfl_xor(s2, 1, 32);
        float mu  = s * (1.0f / 96.0f);
        float var = s2 * (1.0f / 96.0f) - mu * mu;
        float inv = rsqrtf(var + EPSV);
        _Float16* dst = &Abuf[wave][r * 96 + part * 48];
#pragma unroll
        for (int j = 0; j < 48; ++j) {
            int c = part * 48 + j;
            dst[j] = (_Float16)((vals[j] - mu) * inv * lng[c] + lnb[c]);
        }
    }
    lds_fence();

    int hh = lane >> 4, n = lane & 15, rr = lane & 15;
    const _Float16* arow = &Abuf[wave][rr * 96];

    float8 acc[8];
#pragma unroll
    for (int t = 0; t < 8; ++t) acc[t] = fzero8();

#pragma unroll
    for (int kc = 0; kc < 3; ++kc) {
        half16 a;
        half8 lo = *(const half8*)(arow + kc * 32 + 16 * hh);
        half8 hi = *(const half8*)(arow + kc * 32 + 16 * hh + 8);
#pragma unroll
        for (int j = 0; j < 8; ++j) { a[j] = lo[j]; a[8 + j] = hi[j]; }
#pragma unroll
        for (int t = 0; t < 8; ++t) {
            half16 bf = *(const half16*)(wqB + (size_t)((kc * 8 + t) * 32 + lane) * 16);
            acc[t] = __builtin_amdgcn_wmma_f32_16x16x32_f16(
                false, a, false, bf, (short)0, acc[t], false, false);
        }
    }

#pragma unroll
    for (int t = 0; t < 8; ++t) {
        float bias_t = bq[t * 16 + n];
#pragma unroll
        for (int rg = 0; rg < 8; ++rg) {
            int grow = row0 + rg + 8 * hh;
            _Float16 hv = (_Float16)(acc[t][rg] + bias_t);
            if (t == 0) {
                qw[(size_t)grow * 16 + n] = hv;
            } else if (t == 1) {
                kw[(size_t)grow * 16 + n] = hv;
            } else {
                int bb = grow >> 10, nn = grow & 1023;
                int kc2 = nn >> 5, kk2 = nn & 31;
                int hv2 = kk2 >> 4, ee = kk2 & 15;
                int lv = hv2 * 16 + n;
                vB[(((size_t)bb * 32 + kc2) * 6 + (t - 2)) * 512 + lv * 16 + ee] = hv;
            }
        }
    }
}

// ---------------------------------------------------------------------------
// Kernel 2: flash attention. One wave = 16 queries; online softmax over 32
// chunks of 32 keys. d_qk=16 padded to K=32 for the QK WMMA.
// ---------------------------------------------------------------------------
__global__ __launch_bounds__(256) void k_attn(
    const _Float16* __restrict__ qw, const _Float16* __restrict__ kw,
    const _Float16* __restrict__ vB, _Float16* __restrict__ ow) {
    __shared__ __align__(16) _Float16 Pbuf[8][16 * 32];
    int wave = threadIdx.x >> 5, lane = threadIdx.x & 31;
    int b = blockIdx.x >> 3;
    int q0 = (blockIdx.x & 7) * 128 + wave * 16;
    int hh = lane >> 4, n = lane & 15, rr = lane & 15;
    const _Float16 hz = (_Float16)0.f;

    // Q A-fragment (K 16..31 are zero padding)
    half16 qa;
    {
        const _Float16* qbase = qw + ((size_t)b * NTOK + q0) * 16;
        half8 lo = *(const half8*)(qbase + rr * 16);
        half8 hi = *(const half8*)(qbase + rr * 16 + 8);
#pragma unroll
        for (int j = 0; j < 8; ++j) { qa[j] = hh ? hz : lo[j]; qa[8 + j] = hh ? hz : hi[j]; }
    }

    float m[8], l[8];
    float8 o[6];
#pragma unroll
    for (int rg = 0; rg < 8; ++rg) { m[rg] = -1e30f; l[rg] = 0.f; }
#pragma unroll
    for (int t = 0; t < 6; ++t) o[t] = fzero8();

    const _Float16* kbase = kw + (size_t)b * NTOK * 16;
    const _Float16* vbase = vB + (size_t)b * 32 * 6 * 512;
    _Float16* Pw = Pbuf[wave];

    for (int kc = 0; kc < 32; ++kc) {
        int key0 = kc * 32;
        // K B-fragments for two sub-tiles of 16 keys (rows 16..31 = zero pad)
        half16 kb0, kb1;
        if (hh == 0) {
            kb0 = *(const half16*)(kbase + (size_t)(key0 + n) * 16);
            kb1 = *(const half16*)(kbase + (size_t)(key0 + 16 + n) * 16);
        } else {
#pragma unroll
            for (int j = 0; j < 16; ++j) { kb0[j] = hz; kb1[j] = hz; }
        }
        float8 s0 = __builtin_amdgcn_wmma_f32_16x16x32_f16(
            false, qa, false, kb0, (short)0, fzero8(), false, false);
        float8 s1 = __builtin_amdgcn_wmma_f32_16x16x32_f16(
            false, qa, false, kb1, (short)0, fzero8(), false, false);

        lds_fence();
#pragma unroll
        for (int rg = 0; rg < 8; ++rg) {
            float tm = fmaxf(s0[rg], s1[rg]);
            tm = fmaxf(tm, __shfl_xor(tm, 1, 32));
            tm = fmaxf(tm, __shfl_xor(tm, 2, 32));
            tm = fmaxf(tm, __shfl_xor(tm, 4, 32));
            tm = fmaxf(tm, __shfl_xor(tm, 8, 32));
            float mn  = fmaxf(m[rg], tm);
            float fac = __expf(m[rg] - mn);
            m[rg] = mn;
            float p0 = __expf(s0[rg] - mn);
            float p1 = __expf(s1[rg] - mn);
            int row = rg + 8 * hh;
            Pw[row * 32 + n]      = (_Float16)p0;
            Pw[row * 32 + 16 + n] = (_Float16)p1;
            float ps = p0 + p1;
            ps += __shfl_xor(ps, 1, 32);
            ps += __shfl_xor(ps, 2, 32);
            ps += __shfl_xor(ps, 4, 32);
            ps += __shfl_xor(ps, 8, 32);
            l[rg] = l[rg] * fac + ps;
#pragma unroll
            for (int t = 0; t < 6; ++t) o[t][rg] *= fac;
        }
        lds_fence();
        // P A-fragment from LDS
        half16 pa;
        {
            half8 lo = *(const half8*)(Pw + rr * 32 + 16 * hh);
            half8 hi = *(const half8*)(Pw + rr * 32 + 16 * hh + 8);
#pragma unroll
            for (int j = 0; j < 8; ++j) { pa[j] = lo[j]; pa[8 + j] = hi[j]; }
        }
        // P @ V over the 6 dim-tiles; V already in B-fragment order.
#pragma unroll
        for (int t = 0; t < 6; ++t) {
            half16 vb = *(const half16*)(vbase + (size_t)((kc * 6 + t) * 32 + lane) * 16);
            o[t] = __builtin_amdgcn_wmma_f32_16x16x32_f16(
                false, pa, false, vb, (short)0, o[t], false, false);
        }
    }

#pragma unroll
    for (int rg = 0; rg < 8; ++rg) {
        float inv = 1.0f / l[rg];
        size_t grow = (size_t)b * NTOK + q0 + rg + 8 * hh;
#pragma unroll
        for (int t = 0; t < 6; ++t)
            ow[grow * 96 + t * 16 + n] = (_Float16)(o[t][rg] * inv);
    }
}

// ---------------------------------------------------------------------------
// Kernel 3: out = BN(relu(concat(o, x2)) @ proj_w + b) fused. One wave =
// 16 rows x 128 outputs (3 column groups), K = 384 in 12 chunks of 32.
// ---------------------------------------------------------------------------
__global__ __launch_bounds__(128) void k_proj(
    const float* __restrict__ x, const _Float16* __restrict__ ow,
    const _Float16* __restrict__ wpB, const float* __restrict__ bp,
    float* __restrict__ out) {
    int wave = threadIdx.x >> 5, lane = threadIdx.x & 31;
    int wid = blockIdx.x * 4 + wave;
    int rowtile = wid / 3, cg = wid % 3;
    int row0 = rowtile * 16;
    int hh = lane >> 4, n = lane & 15, rr = lane & 15;
    const _Float16 hz = (_Float16)0.f;

    float8 acc[8];
#pragma unroll
    for (int t = 0; t < 8; ++t) acc[t] = fzero8();

    size_t grow_a = (size_t)(row0 + rr);
#pragma unroll
    for (int kc = 0; kc < 12; ++kc) {
        half16 a;
        if (kc < 3) {  // channels 0..95 from attention output, with relu
            half8 lo = *(const half8*)(ow + grow_a * 96 + kc * 32 + 16 * hh);
            half8 hi = *(const half8*)(ow + grow_a * 96 + kc * 32 + 16 * hh + 8);
#pragma unroll
            for (int j = 0; j < 8; ++j) {
                a[j]     = lo[j] > hz ? lo[j] : hz;
                a[8 + j] = hi[j] > hz ? hi[j] : hz;
            }
        } else {       // channels 96..383 from x, with relu
            const float* xb = x + grow_a * DIMC + 96 + (kc - 3) * 32;
            float8 lo = *(const float8*)(xb + 16 * hh);
            float8 hi = *(const float8*)(xb + 16 * hh + 8);
#pragma unroll
            for (int j = 0; j < 8; ++j) {
                a[j]     = (_Float16)fmaxf(lo[j], 0.f);
                a[8 + j] = (_Float16)fmaxf(hi[j], 0.f);
            }
        }
#pragma unroll
        for (int t = 0; t < 8; ++t) {
            half16 bf = *(const half16*)(wpB + (size_t)((kc * 24 + cg * 8 + t) * 32 + lane) * 16);
            acc[t] = __builtin_amdgcn_wmma_f32_16x16x32_f16(
                false, a, false, bf, (short)0, acc[t], false, false);
        }
    }

#pragma unroll
    for (int t = 0; t < 8; ++t) {
        float bias_t = bp[cg * 128 + t * 16 + n];
#pragma unroll
        for (int rg = 0; rg < 8; ++rg) {
            size_t grow = (size_t)(row0 + rg + 8 * hh);
            out[grow * DIMC + cg * 128 + t * 16 + n] = acc[t][rg] + bias_t;
        }
    }
}

// ---------------------------------------------------------------------------
extern "C" void kernel_launch(void* const* d_in, const int* in_sizes, int n_in,
                              void* d_out, int out_size, void* d_ws, size_t ws_size,
                              hipStream_t stream) {
    const float* x     = (const float*)d_in[0];
    const float* ln_g  = (const float*)d_in[1];
    const float* ln_b  = (const float*)d_in[2];
    const float* qkv_w = (const float*)d_in[3];
    const float* qkv_b = (const float*)d_in[4];
    const float* qg    = (const float*)d_in[5];
    const float* qbe   = (const float*)d_in[6];
    const float* qm    = (const float*)d_in[7];
    const float* qv    = (const float*)d_in[8];
    const float* pw    = (const float*)d_in[9];
    const float* pb    = (const float*)d_in[10];
    const float* pg    = (const float*)d_in[11];
    const float* pbe   = (const float*)d_in[12];
    const float* pm    = (const float*)d_in[13];
    const float* pvr   = (const float*)d_in[14];

    char* ws = (char*)d_ws;
    _Float16* wqB = (_Float16*)(ws + 0);                          // 24576 B
    float*    bq  = (float*)(ws + 24576);                         //   512 B
    _Float16* wpB = (_Float16*)(ws + 25088);                      // 294912 B
    float*    bp  = (float*)(ws + 320000);                        //  1536 B
    _Float16* qw  = (_Float16*)(ws + 321536);                     // 2 MB
    _Float16* kw  = (_Float16*)(ws + 321536 + 2097152);           // 2 MB
    _Float16* vB  = (_Float16*)(ws + 321536 + 2 * 2097152);       // 12 MB
    _Float16* ow  = (_Float16*)(ws + 321536 + 2 * 2097152 + 12582912); // 12 MB

    prep_qkv <<<48, 256, 0, stream>>>(qkv_w, qkv_b, qg, qbe, qm, qv, wqB, bq);
    prep_proj<<<576, 256, 0, stream>>>(pw, pb, pg, pbe, pm, pvr, wpB, bp);
    k_qkv    <<<1024, 128, 0, stream>>>(x, ln_g, ln_b, wqB, bq, qw, kw, vB);
    k_attn   <<<512, 256, 0, stream>>>(qw, kw, vB, ow);
    k_proj   <<<3072, 128, 0, stream>>>(x, ow, wpB, bp, (float*)d_out);
}
